// ResidualBlock_7868380086735
// MI455X (gfx1250) — matrix operand order repaired and checked by reading, not compiled, not run on hardware
//
#include <hip/hip_runtime.h>

#define CCH 128            // channels (fixed by problem)
#define TILE_ROWS 128      // rows per workgroup tile
#define LDST 136           // LDS row stride in halves (272 B: 16B-aligned, bank-staggered)

typedef __attribute__((ext_vector_type(16))) __bf16 v16bf;
typedef __attribute__((ext_vector_type(8)))  __bf16 v8bf;
typedef __attribute__((ext_vector_type(8)))  float  v8f;
typedef __attribute__((ext_vector_type(4)))  int    v4i;

#if defined(__gfx1250__) && __has_builtin(__builtin_amdgcn_global_load_async_to_lds_b128) && \
    __has_builtin(__builtin_amdgcn_s_wait_asynccnt)
#define USE_ASYNC_LDS 1
#else
#define USE_ASYNC_LDS 0
#endif

// ---------------- W convert + transpose: [K][cin][cout] f32 -> [K][cout][cin] bf16
__global__ __launch_bounds__(256) void convert_w(const float* __restrict__ W,
                                                 __bf16* __restrict__ Wt, int kcc) {
    int i = blockIdx.x * blockDim.x + threadIdx.x;
    if (i >= kcc) return;
    int k    = i / (CCH * CCH);
    int rem  = i - k * CCH * CCH;
    int cin  = rem >> 7;
    int cout = rem & (CCH - 1);
    Wt[(size_t)k * CCH * CCH + cout * CCH + cin] = (__bf16)W[i];
}

// ---------------- gather -> WMMA GEMM -> scatter-add
// If scale!=null, gathered rows get per-channel relu(v*scale[c]+shift[c]) (fused BN+ReLU).
__global__ __launch_bounds__(256) void sparse_conv_wmma(
    const float* __restrict__ src,      // [n, C] gather source (f32)
    const __bf16* __restrict__ Wt,      // [K][cout][cin] bf16
    const int* __restrict__ idx_in,     // [K, n]
    const int* __restrict__ idx_out,    // [K, n]
    float* __restrict__ dst,            // [n, C] accumulated with atomics
    const float* __restrict__ scale,    // [C] or null
    const float* __restrict__ shift,    // [C] or null
    int n)
{
    // dynamic LDS: Bsm = Wt[k] tile (128 x LDST halves), Asm = gathered rows (128 x LDST)
    extern __shared__ __align__(16) __bf16 smem[];
    __bf16* Bsm = smem;
    __bf16* Asm = smem + CCH * LDST;

    const int k       = blockIdx.y;
    const int rowbase = blockIdx.x * TILE_ROWS;
    const int tid     = threadIdx.x;

    // prefetch scatter indices for this tile (overlaps with staging/compute)
    if (tid < TILE_ROWS / 16)
        __builtin_prefetch(idx_out + (size_t)k * n + rowbase + tid * 16, 0, 1);

    // ---- stage Wt[k] (128x128 bf16) into LDS with padded stride
    const __bf16* wk = Wt + (size_t)k * CCH * CCH;
#pragma unroll
    for (int it = 0; it < 8; ++it) {
        int chunk = tid + it * 256;            // 2048 chunks of 8 halves (16 B)
        int row   = chunk >> 4;
        int co    = (chunk & 15) * 8;
#if USE_ASYNC_LDS
        __builtin_amdgcn_global_load_async_to_lds_b128(
            (__attribute__((address_space(1))) v4i*)(wk + row * CCH + co),
            (__attribute__((address_space(3))) v4i*)(Bsm + row * LDST + co),
            0, 0);
#else
        *(v8bf*)(Bsm + row * LDST + co) = *(const v8bf*)(wk + row * CCH + co);
#endif
    }

    // ---- gather 128 rows (8 threads/row, 16 floats each), f32 -> bf16 into LDS
#pragma unroll
    for (int pass = 0; pass < 4; ++pass) {
        int r    = pass * 32 + (tid >> 3);
        int l8   = tid & 7;
        int grow = rowbase + r;
        v8bf lo = {}, hi = {};
        if (grow < n) {
            int g = idx_in[(size_t)k * n + grow];
            const float4* srow = (const float4*)(src + (size_t)g * CCH + l8 * 16);
            float vals[16];
#pragma unroll
            for (int j = 0; j < 4; ++j) {
                float4 f = srow[j];
                if (scale) {
                    int q = l8 * 4 + j;
                    float4 sc = ((const float4*)scale)[q];
                    float4 sh = ((const float4*)shift)[q];
                    f.x = fmaxf(f.x * sc.x + sh.x, 0.f);
                    f.y = fmaxf(f.y * sc.y + sh.y, 0.f);
                    f.z = fmaxf(f.z * sc.z + sh.z, 0.f);
                    f.w = fmaxf(f.w * sc.w + sh.w, 0.f);
                }
                vals[j * 4 + 0] = f.x; vals[j * 4 + 1] = f.y;
                vals[j * 4 + 2] = f.z; vals[j * 4 + 3] = f.w;
            }
#pragma unroll
            for (int e = 0; e < 8; ++e) {
                lo[e] = (__bf16)vals[e];
                hi[e] = (__bf16)vals[8 + e];
            }
        }
        __bf16* ap = Asm + r * LDST + l8 * 16;
        *(v8bf*)(ap)     = lo;
        *(v8bf*)(ap + 8) = hi;
    }

#if USE_ASYNC_LDS
    __builtin_amdgcn_s_wait_asynccnt(0);   // this wave's async W-tile copies done
#endif
    __syncthreads();                        // all waves' staging visible

    // ---- compute: 8 waves; wave w owns 16-row strip, all 8 column tiles (32 WMMA/wave)
    const int wave  = tid >> 5;
    const int lane  = tid & 31;
    const int r0    = wave * 16;
    const int lrow  = lane & 15;
    const int lhi   = lane >> 4;

    // A fragments (16x32 bf16 each), per ISA layout:
    // lanes 0-15: row=lane, K chunks {0..7, 16..23}; lanes 16-31: row=lane-16, {8..15, 24..31}
    v16bf afrag[4];
#pragma unroll
    for (int ks = 0; ks < 4; ++ks) {
        const __bf16* ab = Asm + (r0 + lrow) * LDST + ks * 32 + lhi * 8;
        v8bf a0 = *(const v8bf*)(ab);
        v8bf a1 = *(const v8bf*)(ab + 16);
        afrag[ks] = __builtin_shufflevector(a0, a1,
            0, 1, 2, 3, 4, 5, 6, 7, 8, 9, 10, 11, 12, 13, 14, 15);
    }

#pragma unroll
    for (int nt = 0; nt < 8; ++nt) {
        v8f acc = {};
#pragma unroll
        for (int ks = 0; ks < 4; ++ks) {
            // B layout: lanes 0-15: col=lane, K {0..7, 16..23}; lanes 16-31: col=lane-16, K {8..15, 24..31}
            const __bf16* bb = Bsm + (nt * 16 + lrow) * LDST + ks * 32 + lhi * 8;
            v8bf b0 = *(const v8bf*)(bb);
            v8bf b1 = *(const v8bf*)(bb + 16);
            v16bf bfrag = __builtin_shufflevector(b0, b1,
                0, 1, 2, 3, 4, 5, 6, 7, 8, 9, 10, 11, 12, 13, 14, 15);
            acc = __builtin_amdgcn_wmma_f32_16x16x32_bf16(
                false, afrag[ks], false, bfrag, (short)0, acc, false, false);
        }
        // scatter: C/D layout: VGPR v -> M = v + 8*(lane>=16); N = lane&15
        int colg = nt * 16 + lrow;
#pragma unroll
        for (int v = 0; v < 8; ++v) {
            int grow = rowbase + r0 + v + lhi * 8;
            if (grow < n) {
                int o = idx_out[(size_t)k * n + grow];
                atomicAdd(dst + (size_t)o * CCH + colg, acc[v]);
            }
        }
    }
}

// ---------------- per-channel sum / sumsq over N rows
__global__ __launch_bounds__(256) void bn_stats(const float* __restrict__ h,
                                                float* __restrict__ sums, int n) {
    __shared__ float ssum[CCH];
    __shared__ float ssq[CCH];
    int tid = threadIdx.x;
    if (tid < CCH) { ssum[tid] = 0.f; ssq[tid] = 0.f; }
    __syncthreads();
    int c = tid & (CCH - 1);
    int r = blockIdx.x * 2 + (tid >> 7);
    int rstep = gridDim.x * 2;
    float s = 0.f, q = 0.f;
    for (; r < n; r += rstep) {
        float v = h[(size_t)r * CCH + c];
        s += v; q += v * v;
    }
    atomicAdd(&ssum[c], s);
    atomicAdd(&ssq[c], q);
    __syncthreads();
    if (tid < CCH) {
        atomicAdd(&sums[tid], ssum[tid]);
        atomicAdd(&sums[CCH + tid], ssq[tid]);
    }
}

// ---------------- mean/var -> per-channel affine (scale, shift)
__global__ void bn_finalize(const float* __restrict__ sums,
                            const float* __restrict__ g, const float* __restrict__ b,
                            float* __restrict__ scale, float* __restrict__ shift, int n) {
    int c = threadIdx.x;
    if (c < CCH) {
        float inv_n = 1.f / (float)n;
        float mu  = sums[c] * inv_n;
        float var = sums[CCH + c] * inv_n - mu * mu;
        float rs  = rsqrtf(var + 1e-5f);
        float sc  = g[c] * rs;
        scale[c] = sc;
        shift[c] = b[c] - mu * sc;
    }
}

// ---------------- out = relu(out*scale[c] + shift[c] + x), in place, float4-wide
__global__ __launch_bounds__(256) void epilogue(float* __restrict__ out,
                                                const float* __restrict__ x,
                                                const float* __restrict__ scale,
                                                const float* __restrict__ shift,
                                                int total4) {
    int i = blockIdx.x * blockDim.x + threadIdx.x;
    int stride = gridDim.x * blockDim.x;
    for (; i < total4; i += stride) {
        int q = i & (CCH / 4 - 1);
        float4 sc = ((const float4*)scale)[q];
        float4 sh = ((const float4*)shift)[q];
        float4 o  = ((const float4*)out)[i];
        float4 xv = ((const float4*)x)[i];
        o.x = fmaxf(o.x * sc.x + sh.x + xv.x, 0.f);
        o.y = fmaxf(o.y * sc.y + sh.y + xv.y, 0.f);
        o.z = fmaxf(o.z * sc.z + sh.z + xv.z, 0.f);
        o.w = fmaxf(o.w * sc.w + sh.w + xv.w, 0.f);
        ((float4*)out)[i] = o;
    }
}

extern "C" void kernel_launch(void* const* d_in, const int* in_sizes, int n_in,
                              void* d_out, int out_size, void* d_ws, size_t ws_size,
                              hipStream_t stream) {
    (void)n_in; (void)out_size; (void)ws_size;
    const float* x      = (const float*)d_in[0];
    const float* W1     = (const float*)d_in[1];
    const float* g1     = (const float*)d_in[2];
    const float* b1     = (const float*)d_in[3];
    const float* W2     = (const float*)d_in[4];
    const float* g2     = (const float*)d_in[5];
    const float* b2     = (const float*)d_in[6];
    const int*   idx_in = (const int*)d_in[7];
    const int*   idx_out= (const int*)d_in[8];
    float* out = (float*)d_out;

    const int n = in_sizes[0] / CCH;                 // 100000
    const int K = in_sizes[1] / (CCH * CCH);         // 27
    const int kcc = K * CCH * CCH;

    // workspace layout
    float* h1     = (float*)d_ws;                    // [n, C]
    float* sums   = h1 + (size_t)n * CCH;            // 4*C: sum1, sq1, sum2, sq2
    float* scale1 = sums + 4 * CCH;
    float* shift1 = scale1 + CCH;
    float* scale2 = shift1 + CCH;
    float* shift2 = scale2 + CCH;
    __bf16* W1t   = (__bf16*)(shift2 + CCH);
    __bf16* W2t   = W1t + (size_t)kcc;

    (void)hipMemsetAsync(h1, 0, (size_t)n * CCH * sizeof(float), stream);
    (void)hipMemsetAsync(out, 0, (size_t)n * CCH * sizeof(float), stream);
    (void)hipMemsetAsync(sums, 0, 4 * CCH * sizeof(float), stream);

    convert_w<<<(kcc + 255) / 256, 256, 0, stream>>>(W1, W1t, kcc);
    convert_w<<<(kcc + 255) / 256, 256, 0, stream>>>(W2, W2t, kcc);

    dim3 cgrid((n + TILE_ROWS - 1) / TILE_ROWS, K);
    const size_t shmem = (size_t)(CCH + TILE_ROWS) * LDST * sizeof(__bf16); // 69,632 B

    // conv1: raw x -> h1
    sparse_conv_wmma<<<cgrid, 256, shmem, stream>>>(x, W1t, idx_in, idx_out, h1,
                                                    nullptr, nullptr, n);
    bn_stats<<<1024, 256, 0, stream>>>(h1, sums, n);
    bn_finalize<<<1, CCH, 0, stream>>>(sums, g1, b1, scale1, shift1, n);

    // conv2: gather applies BN1+ReLU; accumulate straight into d_out
    sparse_conv_wmma<<<cgrid, 256, shmem, stream>>>(h1, W2t, idx_in, idx_out, out,
                                                    scale1, shift1, n);
    bn_stats<<<1024, 256, 0, stream>>>(out, sums + 2 * CCH, n);
    bn_finalize<<<1, CCH, 0, stream>>>(sums + 2 * CCH, g2, b2, scale2, shift2, n);

    // BN2 + residual + ReLU, in place on d_out
    epilogue<<<2048, 256, 0, stream>>>(out, x, scale2, shift2, n * CCH / 4);
}
